// Attention_73744588472852
// MI455X (gfx1250) — matrix operand order repaired and checked by reading, not compiled, not run on hardware
//
#include <hip/hip_runtime.h>
#include <hip/hip_bf16.h>
#include <cstdint>

typedef __attribute__((ext_vector_type(16))) _Float16 v16h;
typedef __attribute__((ext_vector_type(8)))  float    v8f;

#define WMMA_F32_F16(a, b, c) \
  __builtin_amdgcn_wmma_f32_16x16x32_f16(false, (a), false, (b), (short)0, (c), false, false)

constexpr int Bv = 8, Sv = 1374, Dv = 384, Hv = 6, HDv = 64;
constexpr int Sp = 1376;        // padded column count for vT (multiple of 32)
constexpr int Mrows = Bv * Sv;  // 10992

// Async DMA: global -> LDS, 16 bytes per lane, tracked by ASYNCcnt.
__device__ __forceinline__ void async_copy_b128(uint32_t lds_off, const void* gptr) {
  asm volatile("global_load_async_to_lds_b128 %0, %1, off"
               :: "v"(lds_off), "v"((uint64_t)(uintptr_t)gptr)
               : "memory");
}
__device__ __forceinline__ void wait_async_le2() {
  asm volatile("s_wait_asynccnt 2" ::: "memory");
}
__device__ __forceinline__ void wait_async_0() {
  asm volatile("s_wait_asynccnt 0" ::: "memory");
}

// ---------------------------------------------------------------------------
// Weight transpose: wt[n*K + k] = (f16) w[k*N + n]. One-time, tiny (384x384).
// ---------------------------------------------------------------------------
__global__ void transpose_w_f16(const float* __restrict__ w, _Float16* __restrict__ wt,
                                int N, int K) {
  int i = blockIdx.x * blockDim.x + threadIdx.x;
  if (i < N * K) {
    int n = i / K, kk = i - n * K;
    wt[i] = (_Float16)w[(size_t)kk * N + n];
  }
}

__global__ void zero_f32(float* __restrict__ p, int n) {
  int i = blockIdx.x * blockDim.x + threadIdx.x;
  if (i < n) p[i] = 0.f;
}

// Zero the 2 pad columns (s = 1374,1375) of vT so PV tiles load clean zeros.
__global__ void zero_vpad(_Float16* __restrict__ vT) {
  int i = blockIdx.x * blockDim.x + threadIdx.x;
  if (i < Bv * Hv * HDv * (Sp - Sv)) {
    int row = i / (Sp - Sv), c = Sv + (i - row * (Sp - Sv));
    vT[(size_t)row * Sp + c] = (_Float16)0.f;
  }
}

// ---------------------------------------------------------------------------
// LDS-free WMMA GEMM: dst = (A[M,K] * W + bias) * scale, W given pre-transposed
// as f16 Wt[N,K]. All fragments are contiguous 16/32-byte runs per lane
// (global_load_b128 -> v_wmma). Block = 128 threads (4 waves); 64x64 tile.
// TRANS_V: store transposed+padded as dst[((b*H+h)*HD+hd)*Sp + s] (for V).
// ---------------------------------------------------------------------------
template <typename TA, typename TO, bool TRANS_V>
__global__ __launch_bounds__(128) void gemm_wmma(
    const TA* __restrict__ A, const _Float16* __restrict__ Wt,
    const float* __restrict__ bias, TO* __restrict__ dst,
    int M, int N, int K, float scale) {
  const int t = threadIdx.x;
  const int wave = t >> 5, lane = t & 31;
  const int ln = lane & 15;
  const int m0 = blockIdx.x * 64;
  const int n0 = blockIdx.y * 64;

  int arow = m0 + wave * 16 + ln; if (arow >= M) arow = M - 1;
  const TA* arowp = A + (size_t)arow * K + (lane >> 4) * 16;

  v8f acc[4] = {};

  for (int k0 = 0; k0 < K; k0 += 32) {
    // A fragment: 16x32 f16; lane holds row `ln`, K 16*(lane>>4)..+16 contiguous
    v16h af;
    {
      const TA* ap = arowp + k0;
#pragma unroll
      for (int i = 0; i < 8; ++i) {
        af[i]     = (_Float16)ap[i];
        af[8 + i] = (_Float16)ap[8 + i];
      }
    }
#pragma unroll
    for (int ns = 0; ns < 4; ++ns) {
      // B fragment: 32x16; lanes 0-15 hold K 0-15 of column `ln`, lanes 16-31
      // hold K 16-31 -> 16 contiguous halves of Wt row (32B, one b128 pair).
      const _Float16* bp = Wt + (size_t)(n0 + ns * 16 + ln) * K + k0 + (lane >> 4) * 16;
      v16h bf;
#pragma unroll
      for (int i = 0; i < 16; ++i) bf[i] = bp[i];
      acc[ns] = WMMA_F32_F16(af, bf, acc[ns]);
    }
  }

  // Epilogue: C layout -> lanes 0-15: N=lane, M=vgpr; lanes 16-31: N=lane-16, M=8+vgpr
  const int mbase = (lane >> 4) * 8;
#pragma unroll
  for (int ns = 0; ns < 4; ++ns) {
    int n = n0 + ns * 16 + ln;
    float bb = bias[n];
#pragma unroll
    for (int i = 0; i < 8; ++i) {
      int row = m0 + wave * 16 + mbase + i;
      if (row < M) {
        float val = (acc[ns][i] + bb) * scale;
        if (TRANS_V) {
          int bb_ = row / Sv, s = row - bb_ * Sv;
          int hh = n >> 6, hd = n & 63;
          dst[((size_t)((bb_ * Hv + hh) * HDv + hd)) * Sp + s] = (TO)val;
        } else {
          dst[(size_t)row * N + n] = (TO)val;
        }
      }
    }
  }
}

// ---------------------------------------------------------------------------
// Flash-attention pass. Block = 128 threads (4 waves) on one (b,h); each wave
// owns a 16-row Q tile. K/V tiles for each 32-column step are staged into
// double-buffered LDS with async global->LDS DMA (ASYNCcnt), overlapping the
// next tile's copy with the current tile's WMMA + softmax:
//   - logits via 4 WMMAs per wave (q pre-scaled by 1/8)
//   - weighted atomicAdd of unmasked logits into agg[b, col]
//   - for columns < thr: online softmax + P*V via 4 WMMAs (P transposed
//     through per-wave LDS)
// ---------------------------------------------------------------------------
__global__ __launch_bounds__(128) void attn_wmma(
    const _Float16* __restrict__ q, const _Float16* __restrict__ k,
    const _Float16* __restrict__ vT, _Float16* __restrict__ zo,
    float* __restrict__ agg, const int* __restrict__ kp_ptr) {
  __shared__ _Float16 Kt[2][32 * 64];   // [buf][col][e]   4KB each
  __shared__ _Float16 Vt[2][64 * 32];   // [buf][hd][col]  4KB each
  __shared__ _Float16 Plds[4][16 * 32]; // per-wave P transpose staging

  const int t = threadIdx.x, wave = t >> 5, lane = t & 31;
  const int b = blockIdx.z, h = blockIdx.y;
  const int qbase = blockIdx.x * 64 + wave * 16;
  const int thr = kp_ptr[0] + 5;  // columns >= thr are masked (-inf)
  const int ln = lane & 15;
  const int mbase = (lane >> 4) * 8;
  const int eb = (lane >> 4) * 16;
  const size_t vrow = (size_t)((b * Hv + h) * HDv);

  // Q A-fragments for K-dim chunks e=[0,32) and [32,64)
  v16h qa0, qa1;
  {
    int qr = qbase + ln; if (qr >= Sv) qr = Sv - 1;
    const _Float16* qp = q + ((size_t)(b * Sv + qr)) * Dv + h * HDv;
    int kb = (lane >> 4) * 16;
#pragma unroll
    for (int i = 0; i < 8; ++i) {
      qa0[i] = qp[kb + i];      qa0[8 + i] = qp[8 + kb + i];
      qa1[i] = qp[32 + kb + i]; qa1[8 + i] = qp[40 + kb + i];
    }
  }

  // Cooperative async staging of one 32-column step into LDS buffer `buf`.
  auto issue_tiles = [&](int stepi, int buf) {
    int cbl = stepi * 32;
    {  // K tile: 32 cols x 64 halves; thread -> (col, 16-half chunk)
      int col = t >> 2, qd = t & 3;
      int cc = cbl + col; if (cc >= Sv) cc = Sv - 1;
      const _Float16* src = k + ((size_t)(b * Sv + cc)) * Dv + h * HDv + qd * 16;
      async_copy_b128((uint32_t)(uintptr_t)&Kt[buf][col * 64 + qd * 16], src);
    }
    {  // V tile: 64 hd-rows x 32 cols from padded vT; thread -> (hd, chunk)
      int n = t >> 1, qd = t & 1;
      const _Float16* src = vT + (vrow + n) * Sp + cbl + qd * 16;
      async_copy_b128((uint32_t)(uintptr_t)&Vt[buf][n * 32 + qd * 16], src);
    }
  };

  float m_r[8], l_r[8];
#pragma unroll
  for (int i = 0; i < 8; ++i) { m_r[i] = -3.0e38f; l_r[i] = 0.f; }
  v8f o[4] = {};

  const int nsteps = Sp / 32;  // 43 -> columns 0..1375
  issue_tiles(0, 0);

  for (int step = 0; step < nsteps; ++step) {
    const int cb = step * 32;
    const int cur = step & 1;
    const bool have_nxt = (step + 1 < nsteps);

    if (have_nxt) { issue_tiles(step + 1, cur ^ 1); wait_async_le2(); }
    else          { wait_async_0(); }
    __syncthreads();  // current K/V tiles resident for all waves

    // ---- logits: two 16x16 tiles from LDS K ----
    v8f s[2] = {};
#pragma unroll
    for (int hf = 0; hf < 2; ++hf) {
      const _Float16* kp = &Kt[cur][(hf * 16 + ln) * 64 + eb];
      v16h b0, b1;
#pragma unroll
      for (int i = 0; i < 16; ++i) { b0[i] = kp[i]; b1[i] = kp[32 + i]; }
      s[hf] = WMMA_F32_F16(qa0, b0, s[hf]);
      s[hf] = WMMA_F32_F16(qa1, b1, s[hf]);
    }

    // ---- attn_agg: weighted row-sum of unmasked logits ----
#pragma unroll
    for (int hf = 0; hf < 2; ++hf) {
      int col = cb + hf * 16 + ln;
      if (col < Sv) {
        float a = 0.f;
#pragma unroll
        for (int i = 0; i < 8; ++i) {
          int qr = qbase + mbase + i;
          if (qr < Sv) {
            float w = (qr == 0) ? (1.f / 18.f)
                    : (qr <= 4) ? (1.f / 72.f)
                                : (1.f / (18.f * 1369.f));
            a += w * s[hf][i];
          }
        }
        atomicAdd(&agg[(size_t)b * Sv + col], a);
      }
    }

    if (cb < thr) {  // at least one unmasked column in this step
      // ---- mask within step ----
#pragma unroll
      for (int hf = 0; hf < 2; ++hf) {
        int col = cb + hf * 16 + ln;
        if (col >= thr || col >= Sv) {
#pragma unroll
          for (int i = 0; i < 8; ++i) s[hf][i] = -3.0e38f;
        }
      }

      // ---- online softmax (row stats across the 16 column-lanes) ----
      float mx[8];
#pragma unroll
      for (int i = 0; i < 8; ++i) mx[i] = fmaxf(s[0][i], s[1][i]);
#pragma unroll
      for (int off = 1; off < 16; off <<= 1)
#pragma unroll
        for (int i = 0; i < 8; ++i) mx[i] = fmaxf(mx[i], __shfl_xor(mx[i], off, 32));

      float p0[8], p1[8], lsum[8];
#pragma unroll
      for (int i = 0; i < 8; ++i) {
        float mnew = fmaxf(m_r[i], mx[i]);
        float sc = __expf(m_r[i] - mnew);
        m_r[i] = mnew;
        p0[i] = __expf(s[0][i] - mnew);
        p1[i] = __expf(s[1][i] - mnew);
        lsum[i] = p0[i] + p1[i];
        l_r[i] *= sc;
#pragma unroll
        for (int hs = 0; hs < 4; ++hs) o[hs][i] *= sc;
      }
#pragma unroll
      for (int off = 1; off < 16; off <<= 1)
#pragma unroll
        for (int i = 0; i < 8; ++i) lsum[i] += __shfl_xor(lsum[i], off, 32);
#pragma unroll
      for (int i = 0; i < 8; ++i) l_r[i] += lsum[i];

      // ---- transpose P (C layout -> A-fragment) through per-wave LDS ----
      _Float16* pl = &Plds[wave][0];
#pragma unroll
      for (int i = 0; i < 8; ++i) {
        pl[(mbase + i) * 32 + ln]      = (_Float16)p0[i];
        pl[(mbase + i) * 32 + 16 + ln] = (_Float16)p1[i];
      }
      v16h pa;
      {
        const _Float16* pp = &Plds[wave][ln * 32];
        int kb = (lane >> 4) * 16;
#pragma unroll
        for (int i = 0; i < 8; ++i) { pa[i] = pp[kb + i]; pa[8 + i] = pp[8 + kb + i]; }
      }

      // ---- P * V: 4 HD subtiles from LDS V ----
#pragma unroll
      for (int hs = 0; hs < 4; ++hs) {
        const _Float16* vp = &Vt[cur][(hs * 16 + ln) * 32 + eb];
        v16h vb;
#pragma unroll
        for (int i = 0; i < 16; ++i) vb[i] = vp[i];
        o[hs] = WMMA_F32_F16(pa, vb, o[hs]);
      }
    }

    __syncthreads();  // all waves done with buf[cur] before it is overwritten
  }

  // ---- normalize and store zo (f16, [B,S,H,HD] flattened) ----
#pragma unroll
  for (int hs = 0; hs < 4; ++hs) {
#pragma unroll
    for (int i = 0; i < 8; ++i) {
      int qr = qbase + mbase + i;
      if (qr < Sv) {
        float ov = o[hs][i] / l_r[i];
        zo[((size_t)(b * Sv + qr)) * Dv + h * HDv + hs * 16 + ln] = (_Float16)ov;
      }
    }
  }
}

// ---------------------------------------------------------------------------
extern "C" void kernel_launch(void* const* d_in, const int* in_sizes, int n_in,
                              void* d_out, int out_size, void* d_ws, size_t ws_size,
                              hipStream_t stream) {
  const float* x  = (const float*)d_in[0];
  const float* wq = (const float*)d_in[1];
  const float* bq = (const float*)d_in[2];
  const float* wk = (const float*)d_in[3];
  const float* bk = (const float*)d_in[4];
  const float* wv = (const float*)d_in[5];
  const float* bv = (const float*)d_in[6];
  const float* wo = (const float*)d_in[7];  // [H,HD,D] == [384,384] row-major
  const float* bo = (const float*)d_in[8];
  // d_in[9]: keep_patch_indices (unused by the math — only its shape matters)
  const int* kpat = (const int*)d_in[10];   // k_patches scalar

  float* out = (float*)d_out;                       // [B,S,D]
  float* agg = out + (size_t)Mrows * Dv;            // [B,S]

  const size_t NK = (size_t)Dv * Dv;
  _Float16* qf  = (_Float16*)d_ws;
  _Float16* kf  = qf + (size_t)Mrows * Dv;
  _Float16* vT  = kf + (size_t)Mrows * Dv;          // [B*H*HD][Sp]
  _Float16* zf  = vT + (size_t)Bv * Hv * HDv * Sp;
  _Float16* wqt = zf + (size_t)Mrows * Dv;
  _Float16* wkt = wqt + NK;
  _Float16* wvt = wkt + NK;
  _Float16* wot = wvt + NK;

  // One-time weight transposes to f16 [N][K]
  int wtn = (int)NK;
  transpose_w_f16<<<(wtn + 255) / 256, 256, 0, stream>>>(wq, wqt, Dv, Dv);
  transpose_w_f16<<<(wtn + 255) / 256, 256, 0, stream>>>(wk, wkt, Dv, Dv);
  transpose_w_f16<<<(wtn + 255) / 256, 256, 0, stream>>>(wv, wvt, Dv, Dv);
  transpose_w_f16<<<(wtn + 255) / 256, 256, 0, stream>>>(wo, wot, Dv, Dv);

  int npad = Bv * Hv * HDv * (Sp - Sv);
  zero_vpad<<<(npad + 255) / 256, 256, 0, stream>>>(vT);
  zero_f32<<<(Bv * Sv + 255) / 256, 256, 0, stream>>>(agg, Bv * Sv);

  dim3 gp((Mrows + 63) / 64, Dv / 64);  // 172 x 6
  // q pre-scaled by 1/sqrt(HD) = 0.125 (applied to x*Wq + bq, matching reference)
  gemm_wmma<float, _Float16, false><<<gp, 128, 0, stream>>>(x, wqt, bq, qf, Mrows, Dv, Dv, 0.125f);
  gemm_wmma<float, _Float16, false><<<gp, 128, 0, stream>>>(x, wkt, bk, kf, Mrows, Dv, Dv, 1.0f);
  gemm_wmma<float, _Float16, true ><<<gp, 128, 0, stream>>>(x, wvt, bv, vT, Mrows, Dv, Dv, 1.0f);

  dim3 ga((Sv + 63) / 64, Hv, Bv);  // 22 x 6 x 8
  attn_wmma<<<ga, 128, 0, stream>>>(qf, kf, vT, zf, agg, kpat);

  gemm_wmma<_Float16, float, false><<<gp, 128, 0, stream>>>(zf, wot, bo, out, Mrows, Dv, Dv, 1.0f);
}
